// GPTModel_14894946582940
// MI455X (gfx1250) — matrix operand order repaired and checked by reading, not compiled, not run on hardware
//
#include <hip/hip_runtime.h>
#include <hip/hip_bf16.h>

// ---------------------------------------------------------------------------
// GPT forward (4 layers, E=1024, H=16, HD=64, S=2048, FF=4096, V=32000, B=1)
// fp32 residual stream + logits; f16 activations for all WMMA A-operands;
// fp32 weights converted to f16 on the fly while staging B tiles.
// v_wmma_f32_16x16x32_f16 everywhere; async global->LDS (ASYNCcnt) for pure
// f16 tile copies; double-buffered LDS pipeline in the GEMM.
// ---------------------------------------------------------------------------

#define GPT_V  32000
#define GPT_L  4
#define GPT_E  1024
#define GPT_H  16
#define GPT_S  2048
#define GPT_FF 4096
#define GPT_HD 64

typedef _Float16 h2   __attribute__((ext_vector_type(2)));
typedef _Float16 h4   __attribute__((ext_vector_type(4)));
typedef _Float16 h8   __attribute__((ext_vector_type(8)));
typedef _Float16 v16h __attribute__((ext_vector_type(16)));
typedef float    f32x4 __attribute__((ext_vector_type(4)));
typedef float    v8f  __attribute__((ext_vector_type(8)));

static __device__ __forceinline__ v16h combine16(h8 lo, h8 hi) {
    return __builtin_shufflevector(lo, hi, 0,1,2,3,4,5,6,7,8,9,10,11,12,13,14,15);
}

// A fragment (16x32 f16): elems 0..7 <- A[m][half*8+0..7], 8..15 <- A[m][16+half*8+0..7]
static __device__ __forceinline__ v16h load_a_frag(const _Float16* rowPtr, int halfSel) {
    h8 lo = *(const h8*)(rowPtr + halfSel * 8);
    h8 hi = *(const h8*)(rowPtr + 16 + halfSel * 8);
    return combine16(lo, hi);
}

// B fragment (32x16 f16) from transposed LDS tile Bt[n][k]: elems 0..7 <- Bt[n][half*8+0..7], 8..15 <- Bt[n][16+half*8+0..7]
static __device__ __forceinline__ v16h load_b_frag(const _Float16* rowPtr, int halfSel) {
    h8 lo = *(const h8*)(rowPtr + halfSel * 8);
    h8 hi = *(const h8*)(rowPtr + 16 + halfSel * 8);
    return combine16(lo, hi);
}

static __device__ __forceinline__ v8f wmma_f16(v16h a, v16h b, v8f c) {
    return __builtin_amdgcn_wmma_f32_16x16x32_f16(
        false, a, false, b, (short)0, c, false, false);
}

// async global -> LDS copy of 16 bytes per lane (CDNA5 ASYNCcnt path)
static __device__ __forceinline__ void async_copy_b128(const void* gptr, void* lptr) {
    unsigned loff = (unsigned)(unsigned long long)(uintptr_t)lptr; // low 32b = LDS offset
    unsigned long long ga = (unsigned long long)(uintptr_t)gptr;
    asm volatile("global_load_async_to_lds_b128 %0, %1, off"
                 :: "v"(loff), "v"(ga) : "memory");
}
static __device__ __forceinline__ void async_wait0() {
    asm volatile("s_wait_asynccnt 0x0" ::: "memory");
}

static __device__ __forceinline__ float rowReduceMax(float v) {
    #pragma unroll
    for (int m = 1; m < 16; m <<= 1) v = fmaxf(v, __shfl_xor(v, m, 32));
    return v;
}
static __device__ __forceinline__ float rowReduceSum(float v) {
    #pragma unroll
    for (int m = 1; m < 16; m <<= 1) v += __shfl_xor(v, m, 32);
    return v;
}

// ---------------------------------------------------------------------------
// Embedding gather + positional add (fp32 residual stream)
// ---------------------------------------------------------------------------
__global__ __launch_bounds__(256) void embed_kernel(
        const int* __restrict__ x, const float* __restrict__ emb,
        const float* __restrict__ pe, float* __restrict__ h) {
    const int s = blockIdx.x;
    const int tok = x[s];
    const int e = threadIdx.x * 4;                       // E = 256*4
    f32x4 a = *(const f32x4*)&emb[(size_t)tok * GPT_E + e];
    f32x4 p = *(const f32x4*)&pe[(size_t)s * GPT_E + e];
    *(f32x4*)&h[(size_t)s * GPT_E + e] = a + p;
}

// ---------------------------------------------------------------------------
// Row LayerNorm [S,E] fp32 -> f16 (output only ever feeds WMMA A operands)
// ---------------------------------------------------------------------------
__global__ __launch_bounds__(256) void layernorm_kernel(
        const float* __restrict__ x, const float* __restrict__ w,
        const float* __restrict__ b, _Float16* __restrict__ y) {
    const int row = blockIdx.x;
    const int i4 = threadIdx.x * 4;                      // E = 256*4
    const float* xr = x + (size_t)row * GPT_E;
    const f32x4 v = *(const f32x4*)&xr[i4];
    float s1 = v[0] + v[1] + v[2] + v[3];
    float s2 = v[0]*v[0] + v[1]*v[1] + v[2]*v[2] + v[3]*v[3];

    __shared__ float r1[256], r2[256];
    r1[threadIdx.x] = s1; r2[threadIdx.x] = s2;
    __syncthreads();
    for (int off = 128; off > 0; off >>= 1) {
        if (threadIdx.x < off) {
            r1[threadIdx.x] += r1[threadIdx.x + off];
            r2[threadIdx.x] += r2[threadIdx.x + off];
        }
        __syncthreads();
    }
    const float mean = r1[0] * (1.f / GPT_E);
    const float var  = r2[0] * (1.f / GPT_E) - mean * mean;
    const float rstd = rsqrtf(var + 1e-5f);

    const f32x4 wv = *(const f32x4*)&w[i4];
    const f32x4 bv = *(const f32x4*)&b[i4];
    h4 o;
    #pragma unroll
    for (int j = 0; j < 4; ++j)
        o[j] = (_Float16)((v[j] - mean) * rstd * wv[j] + bv[j]);
    *(h4*)&y[(size_t)row * GPT_E + i4] = o;
}

// ---------------------------------------------------------------------------
// WMMA GEMM:  C[M,N] = A(f16)[M,K] @ B(f32->f16)[K,N]  (+bias)(ReLU)(+res)
// 64x128 block tile, K-step 64, double-buffered LDS pipeline:
//   iter i: issue async A(i+1) + global B(i+1) loads, run 8 WMMAs on tile i,
//           cvt/store B(i+1), s_wait_asynccnt, one barrier.
// 8 waves: 16 rows x 64 cols each (4 f32 accumulators).
// Requires M%64==0, N%128==0, K%64==0 (holds for all GEMMs here).
// ---------------------------------------------------------------------------
__global__ __launch_bounds__(256) void gemm_kernel(
        const _Float16* __restrict__ A, const float* __restrict__ B,
        const float* __restrict__ bias, const float* __restrict__ res,
        float* __restrict__ Cf, _Float16* __restrict__ Ch,
        int M, int N, int K, int relu) {
    __shared__ __align__(16) _Float16 As[2][64][72];     // [buf][m][k]
    __shared__ __align__(16) _Float16 Bt[2][128][72];    // [buf][n][k] transposed

    const int tid  = threadIdx.x;
    const int lane = tid & 31;
    const int wid  = tid >> 5;
    const int wm   = wid >> 1;
    const int wn   = wid & 1;
    const int nl   = lane & 15;
    const int halfSel = lane >> 4;
    const int mBase = blockIdx.y * 64;
    const int nBase = blockIdx.x * 128;
    (void)M;

    // fixed per-thread staging coordinates
    const int ar = tid >> 3;              // A row (0..31), +32 for second chunk
    const int ac = (tid & 7) * 8;         // A col chunk (8 halves)
    const int bk = (tid >> 5) * 2;        // B k pair within 16-k slab
    const int bn = (tid & 31) * 4;        // B n quad

    const _Float16* aPtr0 = A + (size_t)(mBase + ar) * K + ac;
    const _Float16* aPtr1 = A + (size_t)(mBase + 32 + ar) * K + ac;
    const float*    bPtr  = B + (size_t)bk * N + nBase + bn;

    v8f acc[4] = {{}, {}, {}, {}};
    f32x4 br[8];

    // ---- prologue: stage tile 0 into buffer 0 ----
    async_copy_b128(aPtr0, &As[0][ar][ac]);
    async_copy_b128(aPtr1, &As[0][32 + ar][ac]);
    #pragma unroll
    for (int rep = 0; rep < 4; ++rep) {
        br[2*rep]     = *(const f32x4*)(bPtr + (size_t)(rep * 16) * N);
        br[2*rep + 1] = *(const f32x4*)(bPtr + (size_t)(rep * 16 + 1) * N);
    }
    #pragma unroll
    for (int rep = 0; rep < 4; ++rep)
        #pragma unroll
        for (int j = 0; j < 4; ++j) {
            h2 p = { (_Float16)br[2*rep][j], (_Float16)br[2*rep + 1][j] };
            *(h2*)&Bt[0][bn + j][rep * 16 + bk] = p;
        }
    async_wait0();
    __syncthreads();

    int cur = 0;
    for (int k0 = 0; k0 < K; k0 += 64) {
        const int nxt = cur ^ 1;
        const bool more = (k0 + 64) < K;

        // ---- issue next tile's loads (async A -> LDS, B -> VGPRs) ----
        if (more) {
            async_copy_b128(aPtr0 + k0 + 64, &As[nxt][ar][ac]);
            async_copy_b128(aPtr1 + k0 + 64, &As[nxt][32 + ar][ac]);
            const float* bp = bPtr + (size_t)(k0 + 64) * N;
            #pragma unroll
            for (int rep = 0; rep < 4; ++rep) {
                br[2*rep]     = *(const f32x4*)(bp + (size_t)(rep * 16) * N);
                br[2*rep + 1] = *(const f32x4*)(bp + (size_t)(rep * 16 + 1) * N);
            }
            if (k0 + 128 < K)
                __builtin_prefetch(bp + (size_t)64 * N, 0, 1);
        }

        // ---- compute on current tile: load all frags, then 4 WMMAs per kc ----
        #pragma unroll
        for (int kcc = 0; kcc < 2; ++kcc) {
            const int kc = kcc * 32;
            const v16h a  = load_a_frag(&As[cur][wm * 16 + nl][kc], halfSel);
            const v16h b0 = load_b_frag(&Bt[cur][wn * 64 +  0 + nl][kc], halfSel);
            const v16h b1 = load_b_frag(&Bt[cur][wn * 64 + 16 + nl][kc], halfSel);
            const v16h b2 = load_b_frag(&Bt[cur][wn * 64 + 32 + nl][kc], halfSel);
            const v16h b3 = load_b_frag(&Bt[cur][wn * 64 + 48 + nl][kc], halfSel);
            acc[0] = wmma_f16(a, b0, acc[0]);
            acc[1] = wmma_f16(a, b1, acc[1]);
            acc[2] = wmma_f16(a, b2, acc[2]);
            acc[3] = wmma_f16(a, b3, acc[3]);
        }

        // ---- finish next tile's staging ----
        if (more) {
            #pragma unroll
            for (int rep = 0; rep < 4; ++rep)
                #pragma unroll
                for (int j = 0; j < 4; ++j) {
                    h2 p = { (_Float16)br[2*rep][j], (_Float16)br[2*rep + 1][j] };
                    *(h2*)&Bt[nxt][bn + j][rep * 16 + bk] = p;
                }
            async_wait0();
        }
        __syncthreads();
        cur = nxt;
    }

    #pragma unroll
    for (int t = 0; t < 4; ++t) {
        #pragma unroll
        for (int r = 0; r < 8; ++r) {
            const int row = mBase + wm * 16 + 8 * halfSel + r;
            const int col = nBase + wn * 64 + t * 16 + nl;
            float v = acc[t][r];
            if (bias) v += bias[col];
            if (relu) v = fmaxf(v, 0.f);
            if (res)  v += res[(size_t)row * N + col];
            if (Ch) Ch[(size_t)row * N + col] = (_Float16)v;
            else    Cf[(size_t)row * N + col] = v;
        }
    }
}

// ---------------------------------------------------------------------------
// Causal flash attention.  Q/K/V f16 [S,E] (head h in columns h*64..h*64+63).
// One block per (64 query rows, head).  1/sqrt(HD) folded into score accums.
// ---------------------------------------------------------------------------
__global__ __launch_bounds__(256) void attn_kernel(
        const _Float16* __restrict__ Q, const _Float16* __restrict__ K,
        const _Float16* __restrict__ Vv, _Float16* __restrict__ O) {
    const int qb = blockIdx.x;
    const int h  = blockIdx.y;
    const int q0 = qb * 64;
    const int ch = h * GPT_HD;

    __shared__ __align__(16) _Float16 Qs[64][72];     // [qrow][d]
    __shared__ __align__(16) _Float16 Ks[64][72];     // [key][d]  (Bt for QK^T)
    __shared__ __align__(16) _Float16 Vt[64][72];     // [d][key]  (Bt for P.V)
    __shared__ __align__(16) _Float16 Ps[4][16][72];  // P tiles (A for P.V)

    const int tid  = threadIdx.x;
    const int lane = tid & 31;
    const int wid  = tid >> 5;
    const int g    = wid >> 1;
    const int dh   = wid & 1;
    const int nl   = lane & 15;
    const int halfSel = lane >> 4;

    const int sr = tid >> 3;              // staging row (0..31)
    const int sc = (tid & 7) * 8;         // staging col (8-half chunk)
    const int vk = (tid >> 3) * 2;        // V transpose: key pair
    const int vd = (tid & 7) * 8;         // V transpose: 8 d values

    async_copy_b128(&Q[(size_t)(q0 + sr) * GPT_E + ch + sc], &Qs[sr][sc]);
    async_copy_b128(&Q[(size_t)(q0 + 32 + sr) * GPT_E + ch + sc], &Qs[32 + sr][sc]);

    v8f o0 = {}, o1 = {};
    float mrow[8], lrow[8];
    #pragma unroll
    for (int r = 0; r < 8; ++r) { mrow[r] = -1e30f; lrow[r] = 0.f; }

    for (int kb = 0; kb <= qb; ++kb) {
        __syncthreads();
        async_copy_b128(&K[(size_t)(kb * 64 + sr) * GPT_E + ch + sc], &Ks[sr][sc]);
        async_copy_b128(&K[(size_t)(kb * 64 + 32 + sr) * GPT_E + ch + sc], &Ks[32 + sr][sc]);
        {
            h8 a = *(const h8*)&Vv[(size_t)(kb * 64 + vk)     * GPT_E + ch + vd];
            h8 b = *(const h8*)&Vv[(size_t)(kb * 64 + vk + 1) * GPT_E + ch + vd];
            #pragma unroll
            for (int j = 0; j < 8; ++j) {
                h2 p = { a[j], b[j] };
                *(h2*)&Vt[vd + j][vk] = p;
            }
        }
        async_wait0();
        __syncthreads();

        // scores: 16x64 = Qg(16x64) @ K^T ; load frags first, then WMMAs
        v8f sA[4] = {{}, {}, {}, {}};
        #pragma unroll
        for (int dcc = 0; dcc < 2; ++dcc) {
            const int dc = dcc * 32;
            const v16h a  = load_a_frag(&Qs[g * 16 + nl][dc], halfSel);
            const v16h b0 = load_b_frag(&Ks[ 0 + nl][dc], halfSel);
            const v16h b1 = load_b_frag(&Ks[16 + nl][dc], halfSel);
            const v16h b2 = load_b_frag(&Ks[32 + nl][dc], halfSel);
            const v16h b3 = load_b_frag(&Ks[48 + nl][dc], halfSel);
            sA[0] = wmma_f16(a, b0, sA[0]);
            sA[1] = wmma_f16(a, b1, sA[1]);
            sA[2] = wmma_f16(a, b2, sA[2]);
            sA[3] = wmma_f16(a, b3, sA[3]);
        }

        const bool diag = (kb == qb);
        #pragma unroll
        for (int r = 0; r < 8; ++r) {
            #pragma unroll
            for (int nt = 0; nt < 4; ++nt) sA[nt][r] *= 0.125f;   // 1/sqrt(HD)
            const int qrow = q0 + g * 16 + 8 * halfSel + r;
            if (diag) {
                #pragma unroll
                for (int nt = 0; nt < 4; ++nt) {
                    const int key = kb * 64 + nt * 16 + nl;
                    if (key > qrow) sA[nt][r] = -1e30f;
                }
            }
            float mx = fmaxf(fmaxf(sA[0][r], sA[1][r]), fmaxf(sA[2][r], sA[3][r]));
            mx = rowReduceMax(mx);
            const float mNew  = fmaxf(mrow[r], mx);
            const float alpha = __expf(mrow[r] - mNew);
            float ssum = 0.f;
            #pragma unroll
            for (int nt = 0; nt < 4; ++nt) {
                const float p = __expf(sA[nt][r] - mNew);
                sA[nt][r] = p;
                ssum += p;
            }
            ssum = rowReduceSum(ssum);
            lrow[r] = lrow[r] * alpha + ssum;
            mrow[r] = mNew;
            o0[r] *= alpha;
            o1[r] *= alpha;
        }

        // write P: the two waves of each row-group write disjoint key halves
        #pragma unroll
        for (int t = 0; t < 2; ++t) {
            const int nt = 2 * dh + t;
            #pragma unroll
            for (int r = 0; r < 8; ++r)
                Ps[g][8 * halfSel + r][nt * 16 + nl] = (_Float16)sA[nt][r];
        }
        __syncthreads();

        // O += P(16x64) @ V(64xd)
        #pragma unroll
        for (int kcc = 0; kcc < 2; ++kcc) {
            const int kc = kcc * 32;
            const v16h a  = load_a_frag(&Ps[g][nl][kc], halfSel);
            const v16h b0 = load_b_frag(&Vt[dh * 32 + nl][kc], halfSel);
            const v16h b1 = load_b_frag(&Vt[dh * 32 + 16 + nl][kc], halfSel);
            o0 = wmma_f16(a, b0, o0);
            o1 = wmma_f16(a, b1, o1);
        }
    }

    #pragma unroll
    for (int r = 0; r < 8; ++r) {
        const float inv = 1.f / lrow[r];
        const int row = q0 + g * 16 + 8 * halfSel + r;
        O[(size_t)row * GPT_E + ch + dh * 32 + nl]      = (_Float16)(o0[r] * inv);
        O[(size_t)row * GPT_E + ch + dh * 32 + 16 + nl] = (_Float16)(o1[r] * inv);
    }
}

// ---------------------------------------------------------------------------
// Host orchestration.  Input order (setup_inputs() insertion order, depth-first):
//   0: x[S]  1: emb[V*E]
//   layer l (base=2+13l): Wq Wk Wv Wo bo ln1_w ln1_b ln2_w ln2_b W1 b1 W2 b2
//   54: lnf_w  55: lnf_b  56: Whead  57: pe
// ---------------------------------------------------------------------------
extern "C" void kernel_launch(void* const* d_in, const int* in_sizes, int n_in,
                              void* d_out, int out_size, void* d_ws, size_t ws_size,
                              hipStream_t stream) {
    (void)in_sizes; (void)n_in; (void)out_size; (void)ws_size;

    const int*   x     = (const int*)d_in[0];
    const float* emb   = (const float*)d_in[1];
    const float* lnf_w = (const float*)d_in[54];
    const float* lnf_b = (const float*)d_in[55];
    const float* Whead = (const float*)d_in[56];
    const float* pe    = (const float*)d_in[57];

    const size_t SE = (size_t)GPT_S * GPT_E;
    float*     hbuf = (float*)d_ws;                 // [S,E] fp32 residual
    _Float16*  th   = (_Float16*)(hbuf + SE);       // LN out (f16)
    _Float16*  qh   = th + SE;
    _Float16*  kh   = qh + SE;
    _Float16*  vh   = kh + SE;
    _Float16*  ah   = vh + SE;                      // attention out (f16)
    _Float16*  fh   = ah + SE;                      // FFN mid [S,FF] (f16)

    const dim3 blk(256);
    const dim3 gEE(GPT_E / 128,  GPT_S / 64);   // 8  x 32
    const dim3 gEF(GPT_FF / 128, GPT_S / 64);   // 32 x 32
    const dim3 gEV(GPT_V / 128,  GPT_S / 64);   // 250 x 32
    const dim3 gAT(GPT_S / 64,   GPT_H);        // 32 x 16

    embed_kernel<<<GPT_S, blk, 0, stream>>>(x, emb, pe, hbuf);

    for (int l = 0; l < GPT_L; ++l) {
        const int base = 2 + 13 * l;
        const float* Wq   = (const float*)d_in[base + 0];
        const float* Wk   = (const float*)d_in[base + 1];
        const float* Wv   = (const float*)d_in[base + 2];
        const float* Wo   = (const float*)d_in[base + 3];
        const float* bo   = (const float*)d_in[base + 4];
        const float* ln1w = (const float*)d_in[base + 5];
        const float* ln1b = (const float*)d_in[base + 6];
        const float* ln2w = (const float*)d_in[base + 7];
        const float* ln2b = (const float*)d_in[base + 8];
        const float* W1   = (const float*)d_in[base + 9];
        const float* b1   = (const float*)d_in[base + 10];
        const float* W2   = (const float*)d_in[base + 11];
        const float* b2   = (const float*)d_in[base + 12];

        layernorm_kernel<<<GPT_S, blk, 0, stream>>>(hbuf, ln1w, ln1b, th);
        gemm_kernel<<<gEE, blk, 0, stream>>>(th, Wq, nullptr, nullptr, nullptr, qh,
                                             GPT_S, GPT_E, GPT_E, 0);
        gemm_kernel<<<gEE, blk, 0, stream>>>(th, Wk, nullptr, nullptr, nullptr, kh,
                                             GPT_S, GPT_E, GPT_E, 0);
        gemm_kernel<<<gEE, blk, 0, stream>>>(th, Wv, nullptr, nullptr, nullptr, vh,
                                             GPT_S, GPT_E, GPT_E, 0);
        attn_kernel<<<gAT, blk, 0, stream>>>(qh, kh, vh, ah);
        gemm_kernel<<<gEE, blk, 0, stream>>>(ah, Wo, bo, hbuf, hbuf, nullptr,
                                             GPT_S, GPT_E, GPT_E, 0);

        layernorm_kernel<<<GPT_S, blk, 0, stream>>>(hbuf, ln2w, ln2b, th);
        gemm_kernel<<<gEF, blk, 0, stream>>>(th, W1, b1, nullptr, nullptr, fh,
                                             GPT_S, GPT_FF, GPT_E, 1 /*relu*/);
        gemm_kernel<<<gEE, blk, 0, stream>>>(fh, W2, b2, hbuf, hbuf, nullptr,
                                             GPT_S, GPT_E, GPT_FF, 0);
    }

    layernorm_kernel<<<GPT_S, blk, 0, stream>>>(hbuf, lnf_w, lnf_b, th);
    gemm_kernel<<<gEV, blk, 0, stream>>>(th, Whead, nullptr, nullptr,
                                         (float*)d_out, nullptr, GPT_S, GPT_V, GPT_E, 0);
}
